// HoloBrainActionDecoder_28707561406912
// MI455X (gfx1250) — matrix operand order repaired and checked by reading, not compiled, not run on hardware
//
#include <hip/hip_runtime.h>
#include <hip/hip_bf16.h>

typedef __bf16 bf16;
typedef __attribute__((ext_vector_type(16))) __bf16 v16bf;
typedef __attribute__((ext_vector_type(8)))  float  v8f;

// ---- model constants ----
#define Bsz   64
#define Jn    7
#define NCn   16
#define Ln    112      // J*NC
#define Dn    512
#define Hn    8
#define DHn   64
#define FFn   2048
#define NLn   6
#define HISTn 4
#define TKn   20       // HIST + NC
#define NIMGn 512
#define PREDn 64
#define FREQn 256

__device__ __forceinline__ float waveReduceSum(float v) {
#pragma unroll
  for (int off = 16; off > 0; off >>= 1) v += __shfl_xor(v, off, 32);
  return v;
}
__device__ __forceinline__ float siluf(float x) { return x / (1.f + __expf(-x)); }

// =====================================================================
// WMMA GEMM:  C[M,N] = A[M,K](bf16, row-major) * Bt[N,K](bf16)^T + bias
// block = 256 threads = 8 waves (2x4); wave tile 32x32 (2x2 WMMA tiles);
// block tile 64x128. Requires M%64==0, N%128==0, K%32==0.
// =====================================================================
union Frag { uint4 u4[2]; v16bf v; };

__device__ __forceinline__ void loadA(Frag& f, const bf16* p, int k0) {
  f.u4[0] = *(const uint4*)(p + k0);
  f.u4[1] = *(const uint4*)(p + k0 + 16);
}
__device__ __forceinline__ void loadB(Frag& f, const bf16* p, int k0) {
  f.u4[0] = *(const uint4*)(p + k0);
  f.u4[1] = *(const uint4*)(p + k0 + 16);
}

__global__ __launch_bounds__(256)
void gemm_bf16_kernel(const bf16* __restrict__ A, const bf16* __restrict__ Bt,
                      const float* __restrict__ bias,
                      float* __restrict__ Cf, bf16* __restrict__ Cb,
                      int M, int N, int K, int act)
{
  const int wave = threadIdx.x >> 5;
  const int lane = threadIdx.x & 31;
  const int m0 = blockIdx.y * 64 + (wave >> 2) * 32;
  const int n0 = blockIdx.x * 128 + (wave & 3) * 32;
  if (m0 >= M || n0 >= N) return;
  const int half = lane >> 4;      // 0 or 1
  const int lid  = lane & 15;
  // A fragment: lanes 0-15 row m, K = {k0..k0+7, k0+16..k0+23}; lanes 16-31: +8
  const bf16* arow0 = A + (size_t)(m0 + lid) * K + half * 8;
  const bf16* arow1 = arow0 + (size_t)16 * K;
  // B fragment: lanes 0-15 col n, K = {k0..k0+7, k0+16..k0+23}; lanes 16-31: +8
  const bf16* brow0 = Bt + (size_t)(n0 + lid) * K + half * 8;
  const bf16* brow1 = brow0 + (size_t)16 * K;
  v8f acc00 = {}, acc01 = {}, acc10 = {}, acc11 = {};
  for (int k0 = 0; k0 < K; k0 += 32) {
    Frag a0, a1, b0, b1;
    loadA(a0, arow0, k0); loadA(a1, arow1, k0);
    loadB(b0, brow0, k0); loadB(b1, brow1, k0);
    acc00 = __builtin_amdgcn_wmma_f32_16x16x32_bf16(false, a0.v, false, b0.v,
                                                    (short)0, acc00, false, false);
    acc01 = __builtin_amdgcn_wmma_f32_16x16x32_bf16(false, a0.v, false, b1.v,
                                                    (short)0, acc01, false, false);
    acc10 = __builtin_amdgcn_wmma_f32_16x16x32_bf16(false, a1.v, false, b0.v,
                                                    (short)0, acc10, false, false);
    acc11 = __builtin_amdgcn_wmma_f32_16x16x32_bf16(false, a1.v, false, b1.v,
                                                    (short)0, acc11, false, false);
  }
  const v8f* accs[4] = { &acc00, &acc01, &acc10, &acc11 };
#pragma unroll
  for (int tile = 0; tile < 4; ++tile) {
    const int mt = m0 + (tile >> 1) * 16;
    const int nt = n0 + (tile & 1) * 16;
    const int n  = nt + lid;
    const float bn = bias ? bias[n] : 0.f;
    const v8f& acc = *accs[tile];
#pragma unroll
    for (int r = 0; r < 8; ++r) {
      const int m = mt + r + half * 8;
      float c = acc[r] + bn;
      if (act) c = siluf(c);
      if (Cf) Cf[(size_t)m * N + n] = c;
      if (Cb) Cb[(size_t)m * N + n] = (bf16)c;
    }
  }
}

// =====================================================================
// Weight convert + transpose:  src[K,N] f32  ->  dst[N,K] bf16
// =====================================================================
__global__ void tcvt_kernel(const float* __restrict__ src, bf16* __restrict__ dst,
                            int K, int N)
{
  long i = (long)blockIdx.x * blockDim.x + threadIdx.x;
  if (i >= (long)K * N) return;
  int k = (int)(i / N), n = (int)(i % N);
  dst[(size_t)n * K + k] = (bf16)src[i];
}

__global__ void cvt_kernel(const float* __restrict__ src, bf16* __restrict__ dst, long n)
{
  long i = (long)blockIdx.x * blockDim.x + threadIdx.x;
  if (i < n) dst[i] = (bf16)src[i];
}

// =====================================================================
// tokenize + embed: (B,PRED,J) -> (B*J*NC, D) bf16 ;  out = x0@w0 + b0
// =====================================================================
__global__ void tokenize_kernel(const float* __restrict__ na, const float* __restrict__ w0,
                                const float* __restrict__ b0, bf16* __restrict__ out)
{
  long i = (long)blockIdx.x * blockDim.x + threadIdx.x;
  if (i >= (long)Bsz * Ln * Dn) return;
  int d = (int)(i % Dn);
  int m = (int)(i / Dn);                // m = (b*J + j)*NC + c
  int c = m % NCn, j = (m / NCn) % Jn, b = m / Ln;
  float acc = b0[d];
#pragma unroll
  for (int ch = 0; ch < 4; ++ch)
    acc += na[((size_t)b * PREDn + c * 4 + ch) * Jn + j] * w0[ch * Dn + d];
  out[i] = (bf16)acc;
}

// =====================================================================
// LayerNorm (wave per row) : in f32 -> out f32 and/or bf16
// =====================================================================
__global__ __launch_bounds__(256)
void ln_kernel(const float* __restrict__ x, const float* __restrict__ w,
               const float* __restrict__ bb, float* __restrict__ outf,
               bf16* __restrict__ outb, int M)
{
  int gw = (int)((blockIdx.x * 256 + threadIdx.x) >> 5);
  int lane = threadIdx.x & 31;
  if (gw >= M) return;
  const float* row = x + (size_t)gw * Dn;
  float s = 0.f, sq = 0.f;
  for (int d = lane; d < Dn; d += 32) { float v = row[d]; s += v; sq += v * v; }
  s = waveReduceSum(s); sq = waveReduceSum(sq);
  float mean = s / Dn, var = sq / Dn - mean * mean;
  float inv = rsqrtf(var + 1e-5f);
  for (int d = lane; d < Dn; d += 32) {
    float y = (row[d] - mean) * inv * w[d] + bb[d];
    if (outf) outf[(size_t)gw * Dn + d] = y;
    if (outb) outb[(size_t)gw * Dn + d] = (bf16)y;
  }
}

// =====================================================================
// RMSNorm (+ optional weight, + optional adaLN shift/scale), wave per row
// =====================================================================
__global__ __launch_bounds__(256)
void rmsmod_kernel(const float* __restrict__ x, const float* __restrict__ w,
                   const float* __restrict__ sh, const float* __restrict__ sc,
                   int modstride, float* __restrict__ outf, bf16* __restrict__ outb,
                   int M, int rowsPerB)
{
  int gw = (int)((blockIdx.x * 256 + threadIdx.x) >> 5);
  int lane = threadIdx.x & 31;
  if (gw >= M) return;
  const float* row = x + (size_t)gw * Dn;
  float ss = 0.f;
  for (int d = lane; d < Dn; d += 32) { float v = row[d]; ss += v * v; }
  ss = waveReduceSum(ss);
  float scale = rsqrtf(ss / Dn + 1e-6f);
  int b = gw / rowsPerB;
  for (int d = lane; d < Dn; d += 32) {
    float y = row[d] * scale;
    if (w)  y *= w[d];
    if (sc) y = y * (1.f + sc[(size_t)b * modstride + d]) + sh[(size_t)b * modstride + d];
    if (outf) outf[(size_t)gw * Dn + d] = y;
    if (outb) outb[(size_t)gw * Dn + d] = (bf16)y;
  }
}

// =====================================================================
// RoPE, rotate-half on each head (D = 8 heads * 64).
// mode 0: tj q  (row=(b*J+j)*NC+c, pos=c+HIST)   mode 1: tj k (pos=row%Tk)
// mode 2: ca q  (pos=(row%L)%NC + 1)             mode 3: img k (pos=row%NIMG)
// =====================================================================
template <typename T>
__global__ void rope_kernel(T* __restrict__ x, int M, int mode)
{
  long idx = (long)blockIdx.x * blockDim.x + threadIdx.x;
  if (idx >= (long)M * Hn * 32) return;
  int i   = (int)(idx & 31);
  int h   = (int)((idx >> 5) & 7);
  int row = (int)(idx >> 8);
  float pos;
  if (mode == 0)      pos = (float)((row % NCn) + HISTn);
  else if (mode == 1) pos = (float)(row % TKn);
  else if (mode == 2) pos = (float)(((row % Ln) % NCn) + 1);
  else                pos = (float)(row % NIMGn);
  float inv = __powf(10000.f, -(2.f * i) / (float)DHn);
  float ang = pos * inv, cs = cosf(ang), sn = sinf(ang);
  size_t base = (size_t)row * Dn + h * DHn;
  float x1 = (float)x[base + i], x2 = (float)x[base + i + 32];
  x[base + i]      = (T)(x1 * cs - x2 * sn);
  x[base + i + 32] = (T)(x2 * cs + x1 * sn);
}

// =====================================================================
// Fused temporal-joint attention: wave per (b, j, c, h); 140 keys.
// =====================================================================
__global__ __launch_bounds__(256)
void attn_tj_kernel(const float* __restrict__ q, const bf16* __restrict__ k,
                    const bf16* __restrict__ v, const float* __restrict__ jrp,
                    const float* __restrict__ slope_raw, bf16* __restrict__ o)
{
  int gw = (int)((blockIdx.x * 256 + threadIdx.x) >> 5);
  int lane = threadIdx.x & 31;
  int h = gw & 7; int t = gw >> 3;
  int c = t % NCn; t /= NCn; int j = t % Jn; int b = t / Jn;
  int qrow = (b * Jn + j) * NCn + c;
  const float* qb = q + (size_t)qrow * Dn + h * DHn;
  float q0 = qb[lane], q1 = qb[lane + 32];
  float slope = log1pf(__expf(slope_raw[h]));
  float mval = -3.4e38f, lsum = 0.f, a0 = 0.f, a1 = 0.f;
  for (int jk = 0; jk < Jn; ++jk) {
    float bias = slope * jrp[j * Jn + jk];
    for (int s = 0; s < TKn; ++s) {
      const bf16* kb = k + ((size_t)(b * Jn + jk) * TKn + s) * Dn + h * DHn;
      float p = q0 * (float)kb[lane] + q1 * (float)kb[lane + 32];
      p = waveReduceSum(p);
      float score = p * 0.125f - bias;
      if (s > c + HISTn) score = -1e9f;
      float mn = fmaxf(mval, score);
      float sc = __expf(mval - mn), e = __expf(score - mn);
      const bf16* vb = v + ((size_t)(b * Jn + jk) * TKn + s) * Dn + h * DHn;
      lsum = lsum * sc + e;
      a0 = a0 * sc + e * (float)vb[lane];
      a1 = a1 * sc + e * (float)vb[lane + 32];
      mval = mn;
    }
  }
  float invl = 1.f / lsum;
  o[(size_t)qrow * Dn + h * DHn + lane]      = (bf16)(a0 * invl);
  o[(size_t)qrow * Dn + h * DHn + lane + 32] = (bf16)(a1 * invl);
}

// =====================================================================
// Fused cross-attention to image features: wave per (b, l, h); 512 keys.
// =====================================================================
__global__ __launch_bounds__(256)
void attn_ca_kernel(const float* __restrict__ q, const bf16* __restrict__ k,
                    const bf16* __restrict__ v, bf16* __restrict__ o)
{
  int gw = (int)((blockIdx.x * 256 + threadIdx.x) >> 5);
  int lane = threadIdx.x & 31;
  int h = gw & 7; int l = (gw >> 3) % Ln; int b = gw / (8 * Ln);
  int qrow = b * Ln + l;
  const float* qb = q + (size_t)qrow * Dn + h * DHn;
  float q0 = qb[lane], q1 = qb[lane + 32];
  float mval = -3.4e38f, lsum = 0.f, a0 = 0.f, a1 = 0.f;
  for (int n = 0; n < NIMGn; ++n) {
    const bf16* kb = k + ((size_t)b * NIMGn + n) * Dn + h * DHn;
    float p = q0 * (float)kb[lane] + q1 * (float)kb[lane + 32];
    p = waveReduceSum(p);
    float score = p * 0.125f;
    float mn = fmaxf(mval, score);
    float sc = __expf(mval - mn), e = __expf(score - mn);
    const bf16* vb = v + ((size_t)b * NIMGn + n) * Dn + h * DHn;
    lsum = lsum * sc + e;
    a0 = a0 * sc + e * (float)vb[lane];
    a1 = a1 * sc + e * (float)vb[lane + 32];
    mval = mn;
  }
  float invl = 1.f / lsum;
  o[(size_t)qrow * Dn + h * DHn + lane]      = (bf16)(a0 * invl);
  o[(size_t)qrow * Dn + h * DHn + lane + 32] = (bf16)(a1 * invl);
}

// =====================================================================
// x = g? g[b,d]*(x+o) : (x+o)
// =====================================================================
__global__ void combine_kernel(float* __restrict__ x, const float* __restrict__ o,
                               const float* __restrict__ g, int modstride,
                               int rowsPerB, long n)
{
  long i = (long)blockIdx.x * blockDim.x + threadIdx.x;
  if (i >= n) return;
  int d = (int)(i % Dn);
  int b = (int)(i / Dn) / rowsPerB;
  float v = x[i] + o[i];
  if (g) v *= g[(size_t)b * modstride + d];
  x[i] = v;
}

// kv = concat(state_features, XF) along time, bf16
__global__ void kvbuild_kernel(const float* __restrict__ sf, const float* __restrict__ xf,
                               bf16* __restrict__ kv)
{
  long i = (long)blockIdx.x * blockDim.x + threadIdx.x;
  if (i >= (long)Bsz * Jn * TKn * Dn) return;
  int d = (int)(i % Dn);
  int s = (int)((i / Dn) % TKn);
  int bj = (int)(i / ((long)Dn * TKn));
  float v = (s < HISTn) ? sf[((size_t)bj * HISTn + s) * Dn + d]
                        : xf[((size_t)bj * NCn + (s - HISTn)) * Dn + d];
  kv[i] = (bf16)v;
}

// timestep embedding -> bf16 (B, 256)
__global__ void temb_kernel(const int* __restrict__ ts, bf16* __restrict__ out)
{
  int i = blockIdx.x * blockDim.x + threadIdx.x;
  if (i >= Bsz * 128) return;
  int b = i / 128, f = i % 128;
  float fr = __expf(-logf(10000.f) * (float)f / 128.f);
  float a = (float)ts[b] * fr;
  out[(size_t)b * FREQn + f]       = (bf16)cosf(a);
  out[(size_t)b * FREQn + 128 + f] = (bf16)sinf(a);
}

__global__ void silucvt_kernel(const float* __restrict__ x, bf16* __restrict__ o, long n)
{
  long i = (long)blockIdx.x * blockDim.x + threadIdx.x;
  if (i < n) o[i] = (bf16)siluf(x[i]);
}

// linear resize along time axis (half-pixel, edge clamp) -> bf16
__global__ void resize_kernel(const float* __restrict__ in, bf16* __restrict__ out,
                              int Tin, int Tout, int BJ)
{
  long i = (long)blockIdx.x * blockDim.x + threadIdx.x;
  if (i >= (long)BJ * Tout * Dn) return;
  int d = (int)(i % Dn);
  int t = (int)((i / Dn) % Tout);
  int r = (int)(i / ((long)Dn * Tout));
  float src = (t + 0.5f) * (float)Tin / (float)Tout - 0.5f;
  float fl = floorf(src), w = src - fl;
  int i0 = (int)fl;
  int ia = min(max(i0, 0), Tin - 1), ib = min(max(i0 + 1, 0), Tin - 1);
  float v = in[((size_t)r * Tin + ia) * Dn + d] * (1.f - w) +
            in[((size_t)r * Tin + ib) * Dn + d] * w;
  out[i] = (bf16)v;
}

// final head: (B*J*PRED, D) bf16 @ out_w1[D] + b -> d_out[b, t, j]
__global__ __launch_bounds__(256)
void head_kernel(const bf16* __restrict__ x, const float* __restrict__ w,
                 const float* __restrict__ bb, float* __restrict__ out)
{
  int gw = (int)((blockIdx.x * 256 + threadIdx.x) >> 5);
  int lane = threadIdx.x & 31;
  const bf16* row = x + (size_t)gw * Dn;
  float s = 0.f;
  for (int d = lane; d < Dn; d += 32) s += (float)row[d] * w[d];
  s = waveReduceSum(s);
  if (lane == 0) {
    int t = gw % PREDn, j = (gw / PREDn) % Jn, b = gw / (PREDn * Jn);
    out[((size_t)b * PREDn + t) * Jn + j] = s + bb[0];
  }
}

// =====================================================================
// Host orchestration
// =====================================================================
extern "C" void kernel_launch(void* const* d_in, const int* in_sizes, int n_in,
                              void* d_out, int out_size, void* d_ws, size_t ws_size,
                              hipStream_t stream)
{
  (void)in_sizes; (void)n_in; (void)out_size; (void)ws_size;
  const float* noisy = (const float*)d_in[0];
  const int*   tsteps = (const int*)d_in[1];
  const float* img   = (const float*)d_in[2];
  const float* statef = (const float*)d_in[3];
  const float* jrp   = (const float*)d_in[4];
  // param flat indices (dict insertion order)
  const int P_IN = 5;          // input block: 14 entries
  const int P_T  = 19;         // t_embed: 4 entries
  const int P_L  = 23;         // layers: 25 entries x 6
  const int P_H  = 173;        // head: 8 entries
  auto F = [&](int i) { return (const float*)d_in[i]; };

  size_t off = 0;
  auto alloc = [&](size_t bytes) -> void* {
    void* p = (char*)d_ws + off;
    off = (off + bytes + 255) & ~(size_t)255;
    return p;
  };
  auto blocks = [](long n, int t) { return (unsigned)((n + t - 1) / t); };

  // ---- bf16 transposed weights ----
  auto tcvt = [&](int idx, int K, int N) -> bf16* {
    bf16* dst = (bf16*)alloc((size_t)K * N * sizeof(bf16));
    tcvt_kernel<<<blocks((long)K * N, 256), 256, 0, stream>>>(F(idx), dst, K, N);
    return dst;
  };
  bf16* w0T   = tcvt(P_IN + 0, 4, Dn);
  (void)w0T;
  bf16* inWT[4];
  inWT[0] = tcvt(P_IN + 2, Dn, Dn);   // w0a
  inWT[1] = tcvt(P_IN + 4, Dn, Dn);   // w0b
  inWT[2] = tcvt(P_IN + 8, Dn, Dn);   // w1a
  inWT[3] = tcvt(P_IN + 10, Dn, Dn);  // w1b
  bf16* ptw1T = tcvt(P_T + 0, FREQn, Dn);
  bf16* ptw2T = tcvt(P_T + 2, Dn, Dn);
  bf16 *adaT[NLn], *tjT[NLn][4], *caT[NLn][4], *f1T[NLn], *f2T[NLn];
  for (int l = 0; l < NLn; ++l) {
    int base = P_L + 25 * l;
    adaT[l] = tcvt(base + 0, Dn, 6 * Dn);
    for (int m = 0; m < 4; ++m) tjT[l][m] = tcvt(base + 2 + m, Dn, Dn);
    for (int m = 0; m < 4; ++m) caT[l][m] = tcvt(base + 10 + m, Dn, Dn);
    f1T[l] = tcvt(base + 21, Dn, FFn);
    f2T[l] = tcvt(base + 23, FFn, Dn);
  }
  bf16* upw0T  = tcvt(P_H + 0, Dn, Dn);
  bf16* upw1T  = tcvt(P_H + 2, Dn, Dn);
  bf16* outw0T = tcvt(P_H + 4, Dn, Dn);

  // ---- activation scratch ----
  const long MX = (long)Bsz * Ln;           // 7168 rows
  const long MK = (long)Bsz * Jn * TKn;     // 8960 rows
  const long MI = (long)Bsz * NIMGn;        // 32768 rows
  float* X    = (float*)alloc(MX * Dn * 4);
  float* XF   = (float*)alloc(MX * Dn * 4);
  bf16*  XFb  = (bf16*)alloc(MX * Dn * 2);
  bf16*  KVb  = (bf16*)alloc(MK * Dn * 2);
  float* Qf   = (float*)alloc(MX * Dn * 4);
  bf16*  Kbig = (bf16*)alloc(MI * Dn * 2);  // also reused as f32 (14336x512)
  bf16*  Vbig = (bf16*)alloc(MI * Dn * 2);
  bf16*  Ob   = (bf16*)alloc(MX * Dn * 2);
  float* OPf  = (float*)alloc(MX * Dn * 4);
  bf16*  Hb   = (bf16*)alloc(MX * FFn * 2);
  bf16*  imgb = (bf16*)alloc(MI * Dn * 2);
  float* mod  = (float*)alloc((size_t)Bsz * 6 * Dn * 4);
  bf16*  Tb   = (bf16*)alloc((size_t)Bsz * FREQn * 2);
  bf16*  TTb  = (bf16*)alloc((size_t)Bsz * Dn * 2);
  float* Tf   = (float*)alloc((size_t)Bsz * Dn * 4);
  bf16*  STb  = (bf16*)alloc((size_t)Bsz * Dn * 2);

  auto gemm = [&](const bf16* A, const bf16* Bt, const float* bias, float* Cf,
                  bf16* Cb, int M, int N, int K, int act) {
    dim3 g(N / 128, M / 64);
    if (g.x == 0) g.x = 1;
    if (g.y == 0) g.y = 1;
    gemm_bf16_kernel<<<g, 256, 0, stream>>>(A, Bt, bias, Cf, Cb, M, N, K, act);
  };

  // img features -> bf16 (reused every layer)
  cvt_kernel<<<blocks(MI * Dn, 256), 256, 0, stream>>>(img, imgb, MI * Dn);

  // ---- input embed stack ----
  tokenize_kernel<<<blocks(MX * Dn, 256), 256, 0, stream>>>(noisy, F(P_IN + 0), F(P_IN + 1), Ob);
  gemm(Ob,  inWT[0], F(P_IN + 3),  nullptr, XFb, (int)MX, Dn, Dn, 1);
  gemm(XFb, inWT[1], F(P_IN + 5),  Qf, nullptr, (int)MX, Dn, Dn, 1);
  ln_kernel<<<blocks(MX, 8) , 256, 0, stream>>>(Qf, F(P_IN + 6), F(P_IN + 7), nullptr, Ob, (int)MX);
  gemm(Ob,  inWT[2], F(P_IN + 9),  nullptr, XFb, (int)MX, Dn, Dn, 1);
  gemm(XFb, inWT[3], F(P_IN + 11), Qf, nullptr, (int)MX, Dn, Dn, 1);
  ln_kernel<<<blocks(MX, 8), 256, 0, stream>>>(Qf, F(P_IN + 12), F(P_IN + 13), X, nullptr, (int)MX);

  // ---- timestep embedder ----
  temb_kernel<<<blocks(Bsz * 128, 256), 256, 0, stream>>>(tsteps, Tb);
  gemm(Tb,  ptw1T, F(P_T + 1), nullptr, TTb, Bsz, Dn, FREQn, 1);
  gemm(TTb, ptw2T, F(P_T + 3), Tf, nullptr, Bsz, Dn, Dn, 0);
  silucvt_kernel<<<blocks((long)Bsz * Dn, 256), 256, 0, stream>>>(Tf, STb, (long)Bsz * Dn);

  const unsigned attnBlocks = (unsigned)(Bsz * Ln * Hn / 8); // 7168 blocks of 8 waves

  // ---- transformer layers ----
  for (int l = 0; l < NLn; ++l) {
    int base = P_L + 25 * l;
    // adaLN modulation
    gemm(STb, adaT[l], F(base + 1), mod, nullptr, Bsz, 6 * Dn, Dn, 0);
    // pre-MSA RMS + shift/scale
    rmsmod_kernel<<<blocks(MX, 8), 256, 0, stream>>>(X, nullptr, mod + 0, mod + Dn,
                                                     6 * Dn, XF, XFb, (int)MX, Ln);
    kvbuild_kernel<<<blocks(MK * Dn, 256), 256, 0, stream>>>(statef, XF, KVb);
    // temporal-joint attention
    gemm(XFb, tjT[l][0], F(base + 6), Qf, nullptr, (int)MX, Dn, Dn, 0);
    rope_kernel<float><<<blocks(MX * 256, 256), 256, 0, stream>>>(Qf, (int)MX, 0);
    gemm(KVb, tjT[l][1], F(base + 7), nullptr, Kbig, (int)MK, Dn, Dn, 0);
    rope_kernel<bf16><<<blocks(MK * 256, 256), 256, 0, stream>>>(Kbig, (int)MK, 1);
    gemm(KVb, tjT[l][2], F(base + 8), nullptr, Vbig, (int)MK, Dn, Dn, 0);
    attn_tj_kernel<<<attnBlocks, 256, 0, stream>>>(Qf, Kbig, Vbig, jrp, F(base + 18), Ob);
    gemm(Ob, tjT[l][3], F(base + 9), OPf, nullptr, (int)MX, Dn, Dn, 0);
    combine_kernel<<<blocks(MX * Dn, 256), 256, 0, stream>>>(X, OPf, mod + 2 * Dn,
                                                             6 * Dn, Ln, MX * Dn);
    // cross-attention to image features
    rmsmod_kernel<<<blocks(MX, 8), 256, 0, stream>>>(X, F(base + 19), nullptr, nullptr,
                                                     0, nullptr, XFb, (int)MX, Ln);
    gemm(XFb, caT[l][0], F(base + 14), Qf, nullptr, (int)MX, Dn, Dn, 0);
    rope_kernel<float><<<blocks(MX * 256, 256), 256, 0, stream>>>(Qf, (int)MX, 2);
    gemm(imgb, caT[l][1], F(base + 15), nullptr, Kbig, (int)MI, Dn, Dn, 0);
    rope_kernel<bf16><<<blocks(MI * 256, 256), 256, 0, stream>>>(Kbig, (int)MI, 3);
    gemm(imgb, caT[l][2], F(base + 16), nullptr, Vbig, (int)MI, Dn, Dn, 0);
    attn_ca_kernel<<<attnBlocks, 256, 0, stream>>>(Qf, Kbig, Vbig, Ob);
    gemm(Ob, caT[l][3], F(base + 17), OPf, nullptr, (int)MX, Dn, Dn, 0);
    combine_kernel<<<blocks(MX * Dn, 256), 256, 0, stream>>>(X, OPf, nullptr, 0, Ln, MX * Dn);
    // FFN with adaLN shift/scale + gate
    rmsmod_kernel<<<blocks(MX, 8), 256, 0, stream>>>(X, F(base + 20), mod + 3 * Dn,
                                                     mod + 4 * Dn, 6 * Dn, nullptr, XFb,
                                                     (int)MX, Ln);
    gemm(XFb, f1T[l], F(base + 22), nullptr, Hb, (int)MX, FFn, Dn, 1);
    gemm(Hb,  f2T[l], F(base + 24), OPf, nullptr, (int)MX, Dn, FFn, 0);
    combine_kernel<<<blocks(MX * Dn, 256), 256, 0, stream>>>(X, OPf, mod + 5 * Dn,
                                                             6 * Dn, Ln, MX * Dn);
  }

  // ---- upsample head ----
  const int BJ = Bsz * Jn;                    // 448
  const long M32 = (long)BJ * 32, M64 = (long)BJ * PREDn;
  float* U0 = (float*)Kbig;                   // reuse (29.4MB fits)
  resize_kernel<<<blocks(M32 * Dn, 256), 256, 0, stream>>>(X, Hb, NCn, 32, BJ);
  gemm(Hb, upw0T, F(P_H + 1), U0, nullptr, (int)M32, Dn, Dn, 1);
  resize_kernel<<<blocks(M64 * Dn, 256), 256, 0, stream>>>(U0, Hb, 32, PREDn, BJ);
  gemm(Hb, upw1T, F(P_H + 3), nullptr, Vbig, (int)M64, Dn, Dn, 1);
  gemm(Vbig, outw0T, F(P_H + 5), nullptr, Hb, (int)M64, Dn, Dn, 1);
  head_kernel<<<blocks(M64, 8), 256, 0, stream>>>(Hb, F(P_H + 6), F(P_H + 7), (float*)d_out);
}
